// LocalEncoder_62448824484365
// MI455X (gfx1250) — matrix operand order repaired and checked by reading, not compiled, not run on hardware
//
#include <hip/hip_runtime.h>
#include <hip/hip_bf16.h>

#define B_  4
#define S_  2048
#define D_  512
#define L_  6
#define H_  8
#define DH_ 64
#define F_  2048
#define V_  260
#define P_  256
#define G_  768
#define EPS_ 1e-3f

#ifndef USE_TDM
#define USE_TDM 1
#endif

typedef __attribute__((ext_vector_type(16))) __bf16 v16bf;
typedef __attribute__((ext_vector_type(8)))  __bf16 v8bf;
typedef __attribute__((ext_vector_type(8)))  float  v8f;
typedef __attribute__((ext_vector_type(4)))  unsigned int u32x4;
typedef __attribute__((ext_vector_type(8)))  int i32x8;
typedef __attribute__((ext_vector_type(4)))  int i32x4;

static __device__ __forceinline__ v8f vzero8() {
  v8f z;
#pragma unroll
  for (int i = 0; i < 8; ++i) z[i] = 0.0f;
  return z;
}

static __device__ __forceinline__ v16bf frag2(const __bf16* lo, const __bf16* hi) {
  v8bf a = *(const v8bf*)lo;
  v8bf b = *(const v8bf*)hi;
  v16bf r;
#pragma unroll
  for (int i = 0; i < 8; ++i) { r[i] = a[i]; r[i + 8] = b[i]; }
  return r;
}

// Branch-free tanh-approximate GeLU (matches jax.nn.gelu approximate=True).
static __device__ __forceinline__ float gelu_tanh(float u) {
  float z  = 0.7978845608028654f * (u + 0.044715f * u * u * u);
  float az = fabsf(z);
  float e  = __expf(-2.0f * az);
  float th = (1.0f - e) / (1.0f + e);
  th = copysignf(th, z);
  return 0.5f * u * (1.0f + th);
}

#if USE_TDM
// TDM 2D tile load: global bf16 (row-major, row_stride halves) -> LDS at
// lds_off. Rows of tile_k halves, LDS-padded by 4 DWORDs (8 halves) every
// 2^(pad_code+1) DWORDs; pad_code=3 -> 32-half rows padded to 40,
// pad_code=4 -> 64-half rows padded to 72.
static __device__ __forceinline__ void tdm_load_tile(
    unsigned lds_off, const void* gaddr, unsigned tile_k_halves,
    unsigned tile_rows, unsigned row_stride_halves, unsigned pad_code) {
  unsigned long long ga = (unsigned long long)(size_t)gaddr;
  u32x4 g0;
  g0[0] = 1u;                                   // count=1 (valid user descriptor)
  g0[1] = lds_off;                              // lds_addr (bytes)
  g0[2] = (unsigned)(ga & 0xFFFFFFFFu);         // global_addr[31:0]
  g0[3] = (unsigned)((ga >> 32) & 0x01FFFFFFu) | (2u << 30);  // addr[56:32] | type=2
  i32x8 g1;
  g1[0] = (int)((1u << 16) | (1u << 20) | (pad_code << 22) | (3u << 25)); // 2B elems, pad_en, interval, amount=4dw
  g1[1] = (int)((tile_k_halves & 0xFFFFu) << 16);                   // tensor_dim0 lo16
  g1[2] = (int)(((tile_k_halves >> 16) & 0xFFFFu) | ((tile_rows & 0xFFFFu) << 16)); // dim0 hi | dim1 lo
  g1[3] = (int)(((tile_rows >> 16) & 0xFFFFu) | ((tile_k_halves & 0xFFFFu) << 16)); // dim1 hi | tile_dim0
  g1[4] = (int)(tile_rows & 0xFFFFu);                               // tile_dim1 (tile_dim2=0)
  g1[5] = (int)row_stride_halves;                                   // tensor_dim0_stride lo32
  g1[6] = 0;
  g1[7] = 0;
  i32x4 z4;
  z4[0] = 0; z4[1] = 0; z4[2] = 0; z4[3] = 0;
#if defined(__clang_major__) && (__clang_major__ >= 23)
  i32x8 z8;
#pragma unroll
  for (int i = 0; i < 8; ++i) z8[i] = 0;
  __builtin_amdgcn_tensor_load_to_lds(g0, g1, z4, z4, z8, 0);
#else
  __builtin_amdgcn_tensor_load_to_lds(g0, g1, z4, z4, 0);
#endif
}
#endif

// ---------------------------------------------------------------------------
// GEMM: C[M,N] = act(A[M,K] @ Wt[N,K]^T + bias) (+Res). A,Wt bf16; acc f32.
// Block tile 64x256, BK=32; 8 waves each 32x64 (8 v_wmma per k-chunk).
// TDM double-buffered: chunk c+1 DMA overlaps chunk c WMMAs.
// Requires M%64==0, N%256==0, K%32==0 (true for all shapes here).
// ---------------------------------------------------------------------------
__global__ __launch_bounds__(256)
void gemm_kernel(const __bf16* __restrict__ A, const __bf16* __restrict__ Wt,
                 const float* __restrict__ bias, const float* __restrict__ Res,
                 float* __restrict__ Cf, __bf16* __restrict__ Cb,
                 int M, int N, int K, int act) {
  __shared__ __bf16 As[2 * 64 * 40];    // [buf][m][k] padded rows
  __shared__ __bf16 Bs[2 * 256 * 40];   // [buf][n][k] padded rows

  const int t    = threadIdx.x;
  const int lane = t & 31;
  const int w    = t >> 5;
  const int wm   = w >> 2;          // 0..1
  const int wn   = w & 3;           // 0..3
  const int bm   = blockIdx.y * 64;
  const int bn   = blockIdx.x * 256;
  const int l15  = lane & 15;
  const int lh   = lane >> 4;

  v8f acc[2][4];
#pragma unroll
  for (int i = 0; i < 2; ++i)
#pragma unroll
    for (int j = 0; j < 4; ++j) acc[i][j] = vzero8();

  const int nch = K >> 5;
#if USE_TDM
  const unsigned as_base = (unsigned)(size_t)(void*)As;
  const unsigned bs_base = (unsigned)(size_t)(void*)Bs;
  if (w == 0) {
    tdm_load_tile(as_base, A + (size_t)bm * K, 32, 64, (unsigned)K, 3);
    tdm_load_tile(bs_base, Wt + (size_t)bn * K, 32, 256, (unsigned)K, 3);
    __builtin_amdgcn_s_wait_tensorcnt(0);
  }
  __syncthreads();
#endif

  for (int c = 0; c < nch; ++c) {
    const int cur = c & 1;
#if USE_TDM
    if (w == 0 && c + 1 < nch) {
      const unsigned nb = (unsigned)(1 - cur);
      tdm_load_tile(as_base + nb * (64 * 40 * 2),
                    A + (size_t)bm * K + (c + 1) * 32, 32, 64, (unsigned)K, 3);
      tdm_load_tile(bs_base + nb * (256 * 40 * 2),
                    Wt + (size_t)bn * K + (c + 1) * 32, 32, 256, (unsigned)K, 3);
    }
    const __bf16* Ac = As + cur * (64 * 40);
    const __bf16* Bc = Bs + cur * (256 * 40);
#else
    __syncthreads();
    {
      const int row = t >> 2, seg = (t & 3) * 8;
      *(v8bf*)(As + row * 40 + seg) = *(const v8bf*)(A + (size_t)(bm + row) * K + c * 32 + seg);
      const __bf16* wp = Wt + (size_t)(bn + t) * K + c * 32;
      __bf16* bs = Bs + t * 40;
#pragma unroll
      for (int i = 0; i < 4; ++i)
        *(v8bf*)(bs + i * 8) = *(const v8bf*)(wp + i * 8);
    }
    __syncthreads();
    const __bf16* Ac = As;
    const __bf16* Bc = Bs;
#endif

    v16bf af[2], bf[4];
#pragma unroll
    for (int mt = 0; mt < 2; ++mt) {
      const __bf16* ab = Ac + (wm * 32 + mt * 16 + l15) * 40 + lh * 8;
      af[mt] = frag2(ab, ab + 16);
    }
#pragma unroll
    for (int nt = 0; nt < 4; ++nt) {
      const __bf16* bb = Bc + (wn * 64 + nt * 16 + l15) * 40 + lh * 8;
      bf[nt] = frag2(bb, bb + 16);
    }
#pragma unroll
    for (int mt = 0; mt < 2; ++mt)
#pragma unroll
      for (int nt = 0; nt < 4; ++nt)
        acc[mt][nt] = __builtin_amdgcn_wmma_f32_16x16x32_bf16(
            false, af[mt], false, bf[nt], (short)0, acc[mt][nt], false, false);

#if USE_TDM
    if (w == 0) __builtin_amdgcn_s_wait_tensorcnt(0);
    __syncthreads();
#endif
  }

#pragma unroll
  for (int mt = 0; mt < 2; ++mt)
#pragma unroll
    for (int nt = 0; nt < 4; ++nt) {
      int col = bn + wn * 64 + nt * 16 + l15;
      int row0 = bm + wm * 32 + mt * 16 + (lh << 3);
      float bv = bias[col];
      float rv[8];
      if (Res) {
#pragma unroll
        for (int r = 0; r < 8; ++r) rv[r] = Res[(size_t)(row0 + r) * N + col];
      }
#pragma unroll
      for (int r = 0; r < 8; ++r) {
        float v = acc[mt][nt][r] + bv;
        if (act == 1) v = gelu_tanh(v);
        if (Res) v += rv[r];
        if (Cf) Cf[(size_t)(row0 + r) * N + col] = v;
        if (Cb) Cb[(size_t)(row0 + r) * N + col] = (__bf16)v;
      }
    }
}

// ---------------------------------------------------------------------------
// Flash attention over bf16 qkv (score scale pre-folded into K weights).
// Workgroup = (b,h,128 queries); 8 waves x 16 queries; 32-key chunks,
// double-buffered K (TDM) + V (manual transpose) staging.
// ---------------------------------------------------------------------------
__global__ __launch_bounds__(256)
void attn_kernel(const __bf16* __restrict__ qkv, __bf16* __restrict__ o) {
  __shared__ __bf16 Kl[2 * 32 * 72];    // [buf][key][dh]
  __shared__ __bf16 Vl[2 * 64 * 40];    // [buf][dh][key] transposed
  __shared__ __bf16 Ql[8 * 16 * 72];    // per-wave [q][dh]
  __shared__ __bf16 Pl[8 * 16 * 40];    // per-wave [q][key32]

  const int t    = threadIdx.x;
  const int lane = t & 31;
  const int w    = t >> 5;
  const int l15  = lane & 15;
  const int lh   = lane >> 4;
  const int b    = blockIdx.z;
  const int h    = blockIdx.y;
  const int q0   = blockIdx.x * 128;
  const int RS   = 3 * D_;  // 1536
  const int kp_t = t & 15;          // key pair (V staging)
  const int dsg  = (t >> 4) * 4;    // dh segment (V staging)

  // Stage V chunk into Vl[buf]: transpose [32 key][64 dh] -> [dh][key],
  // packing key pairs into b32 LDS stores.
  auto stage_v = [&](int kb, int buf) {
    const unsigned short* u0 =
        (const unsigned short*)(qkv + (size_t)(b * S_ + kb + 2 * kp_t) * RS + 2 * D_ + h * 64 + dsg);
    const unsigned short* u1 = u0 + RS;
    unsigned* Vw = (unsigned*)(Vl + buf * (64 * 40));
#pragma unroll
    for (int i = 0; i < 4; ++i)
      Vw[(dsg + i) * 20 + kp_t] = (unsigned)u0[i] | ((unsigned)u1[i] << 16);
  };

#if USE_TDM
  const unsigned kl_base = (unsigned)(size_t)(void*)Kl;
  // Per-wave Q tile via TDM (each wave waits only on its own TENSORcnt).
  tdm_load_tile((unsigned)(size_t)(void*)Ql + (unsigned)(w * 16 * 72 * 2),
                qkv + (size_t)(b * S_ + q0 + w * 16) * RS + h * 64, 64, 16, (unsigned)RS, 4);
  if (w == 0)
    tdm_load_tile(kl_base, qkv + (size_t)(b * S_) * RS + D_ + h * 64, 64, 32, (unsigned)RS, 4);
  __builtin_amdgcn_s_wait_tensorcnt(0);
#else
  {
    const __bf16* qp = qkv + (size_t)(b * S_ + q0 + w * 16 + l15) * RS + h * 64 + lh * 32;
    __bf16* qs = Ql + (w * 16 + l15) * 72 + lh * 32;
#pragma unroll
    for (int i = 0; i < 4; ++i)
      *(v8bf*)(qs + i * 8) = *(const v8bf*)(qp + i * 8);
  }
  {
    const int key = t >> 3, seg = (t & 7) * 8;
    *(v8bf*)(Kl + key * 72 + seg) =
        *(const v8bf*)(qkv + (size_t)(b * S_ + key) * RS + D_ + h * 64 + seg);
  }
#endif
  stage_v(0, 0);

  v16bf aq0, aq1;
  {
    const __bf16* base = Ql + (w * 16 + l15) * 72 + lh * 8;
    aq0 = frag2(base, base + 16);
    aq1 = frag2(base + 32, base + 48);
  }

  float m_run[8], l_run[8];
  v8f acc_o[4];
#pragma unroll
  for (int r = 0; r < 8; ++r) { m_run[r] = -INFINITY; l_run[r] = 0.0f; }
#pragma unroll
  for (int tt = 0; tt < 4; ++tt) acc_o[tt] = vzero8();

  __syncthreads();

  const int NCH = S_ / 32;  // 64
  for (int c = 0; c < NCH; ++c) {
    const int cur = c & 1;
    if (c + 1 < NCH) {
      const int nxt = 1 - cur;
#if USE_TDM
      if (w == 0)
        tdm_load_tile(kl_base + (unsigned)nxt * (32 * 72 * 2),
                      qkv + (size_t)(b * S_ + (c + 1) * 32) * RS + D_ + h * 64,
                      64, 32, (unsigned)RS, 4);
#else
      const int key = t >> 3, seg = (t & 7) * 8;
      *(v8bf*)(Kl + nxt * (32 * 72) + key * 72 + seg) =
          *(const v8bf*)(qkv + (size_t)(b * S_ + (c + 1) * 32 + key) * RS + D_ + h * 64 + seg);
#endif
      stage_v((c + 1) * 32, nxt);
    }
    const __bf16* Kc = Kl + cur * (32 * 72);
    const __bf16* Vc = Vl + cur * (64 * 40);

    // Scores: 16x32 = Q(16x64) . K^T
    v8f sc[2];
#pragma unroll
    for (int nt = 0; nt < 2; ++nt) {
      sc[nt] = vzero8();
      const __bf16* b0 = Kc + (nt * 16 + l15) * 72 + lh * 8;
      v16bf k0f = frag2(b0, b0 + 16);
      sc[nt] = __builtin_amdgcn_wmma_f32_16x16x32_bf16(false, aq0, false, k0f, (short)0, sc[nt], false, false);
      v16bf k1f = frag2(b0 + 32, b0 + 48);
      sc[nt] = __builtin_amdgcn_wmma_f32_16x16x32_bf16(false, aq1, false, k1f, (short)0, sc[nt], false, false);
    }

    // Online softmax
#pragma unroll
    for (int r = 0; r < 8; ++r) {
      float cm = fmaxf(sc[0][r], sc[1][r]);
#pragma unroll
      for (int off = 8; off >= 1; off >>= 1) cm = fmaxf(cm, __shfl_xor(cm, off, 32));
      float mnew  = fmaxf(m_run[r], cm);
      float alpha = __expf(m_run[r] - mnew);
      float p0 = __expf(sc[0][r] - mnew);
      float p1 = __expf(sc[1][r] - mnew);
      float ls = p0 + p1;
#pragma unroll
      for (int off = 8; off >= 1; off >>= 1) ls += __shfl_xor(ls, off, 32);
      l_run[r] = l_run[r] * alpha + ls;
      m_run[r] = mnew;
      __bf16* pr = Pl + (w * 16 + r + (lh << 3)) * 40 + l15;
      pr[0]  = (__bf16)p0;
      pr[16] = (__bf16)p1;
#pragma unroll
      for (int tt = 0; tt < 4; ++tt) acc_o[tt][r] *= alpha;
    }

    // O += P(16x32) . V(32x64)
    const __bf16* pb = Pl + (w * 16 + l15) * 40 + lh * 8;
    v16bf pf = frag2(pb, pb + 16);
#pragma unroll
    for (int tt = 0; tt < 4; ++tt) {
      const __bf16* vb = Vc + (tt * 16 + l15) * 40 + lh * 8;
      v16bf vf = frag2(vb, vb + 16);
      acc_o[tt] = __builtin_amdgcn_wmma_f32_16x16x32_bf16(false, pf, false, vf, (short)0, acc_o[tt], false, false);
    }

#if USE_TDM
    if (w == 0) __builtin_amdgcn_s_wait_tensorcnt(0);
#endif
    __syncthreads();
  }

#pragma unroll
  for (int r = 0; r < 8; ++r) {
    float inv = 1.0f / l_run[r];
    int q = q0 + w * 16 + r + (lh << 3);
#pragma unroll
    for (int tt = 0; tt < 4; ++tt)
      o[(size_t)(b * S_ + q) * D_ + h * 64 + tt * 16 + l15] = (__bf16)(acc_o[tt][r] * inv);
  }
}

// ---------------------------------------------------------------------------
// LayerNorm over D=512; writes f32 and optional bf16 copy.
// ---------------------------------------------------------------------------
__global__ __launch_bounds__(256)
void ln_kernel(const float* __restrict__ in, const float* __restrict__ g,
               const float* __restrict__ bta, float* __restrict__ out,
               __bf16* __restrict__ out_bf) {
  __shared__ float red[256];
  const int row = blockIdx.x;
  const int t   = threadIdx.x;
  const float* p = in + (size_t)row * D_;
  float v0 = p[t], v1 = p[t + 256];
  red[t] = v0 + v1;
  __syncthreads();
#pragma unroll
  for (int o = 128; o > 0; o >>= 1) { if (t < o) red[t] += red[t + o]; __syncthreads(); }
  float mean = red[0] * (1.0f / 512.0f);
  __syncthreads();
  float d0 = v0 - mean, d1 = v1 - mean;
  red[t] = d0 * d0 + d1 * d1;
  __syncthreads();
#pragma unroll
  for (int o = 128; o > 0; o >>= 1) { if (t < o) red[t] += red[t + o]; __syncthreads(); }
  float rs = rsqrtf(red[0] * (1.0f / 512.0f) + EPS_);
  float r0 = d0 * rs * g[t] + bta[t];
  float r1 = d1 * rs * g[t + 256] + bta[t + 256];
  out[(size_t)row * D_ + t]       = r0;
  out[(size_t)row * D_ + t + 256] = r1;
  if (out_bf) {
    out_bf[(size_t)row * D_ + t]       = (__bf16)r0;
    out_bf[(size_t)row * D_ + t + 256] = (__bf16)r1;
  }
}

__global__ __launch_bounds__(256)
void embed_kernel(const int* __restrict__ tok, const float* __restrict__ emb,
                  const float* __restrict__ pos, float* __restrict__ x,
                  __bf16* __restrict__ x_bf) {
  const int bs = blockIdx.x;
  const int s  = bs % S_;
  const int t  = threadIdx.x;
  const int tk = tok[bs];
  float a = emb[(size_t)tk * D_ + t]       + pos[(size_t)s * D_ + t];
  float b = emb[(size_t)tk * D_ + t + 256] + pos[(size_t)s * D_ + t + 256];
  x[(size_t)bs * D_ + t]        = a;
  x[(size_t)bs * D_ + t + 256]  = b;
  x_bf[(size_t)bs * D_ + t]       = (__bf16)a;
  x_bf[(size_t)bs * D_ + t + 256] = (__bf16)b;
}

// Transpose+convert W[K,N] f32 -> Wt[N,K] bf16, scaling columns in [slo,shi).
__global__ __launch_bounds__(256)
void wconv_kernel(const float* __restrict__ W, __bf16* __restrict__ Wt,
                  int K, int N, int slo, int shi, float scale) {
  __shared__ float ts[32][33];
  const int k0 = blockIdx.y * 32, n0 = blockIdx.x * 32;
  const int t  = threadIdx.x;
  {
    int kk = t >> 3, nn4 = (t & 7) * 4;
    float4 v = *(const float4*)(W + (size_t)(k0 + kk) * N + n0 + nn4);
    ts[kk][nn4 + 0] = v.x; ts[kk][nn4 + 1] = v.y;
    ts[kk][nn4 + 2] = v.z; ts[kk][nn4 + 3] = v.w;
  }
  __syncthreads();
  {
    int nn = t >> 3, kk4 = (t & 7) * 4;
    int gn = n0 + nn;
    float s = (gn >= slo && gn < shi) ? scale : 1.0f;
    __bf16* dst = Wt + (size_t)gn * K + k0 + kk4;
#pragma unroll
    for (int i = 0; i < 4; ++i) dst[i] = (__bf16)(ts[kk4 + i][nn] * s);
  }
}

// Scale the K-third of qkv biases by 0.125 (matches score scaling).
__global__ void bscale_kernel(const float* __restrict__ in, float* __restrict__ out, int n) {
  int i = blockIdx.x * 256 + threadIdx.x;
  if (i < n) {
    int col = i % (3 * D_);
    out[i] = in[i] * ((col >= D_ && col < 2 * D_) ? 0.125f : 1.0f);
  }
}

__global__ void zero_kernel(float* __restrict__ p, int n) {
  int i = blockIdx.x * 256 + threadIdx.x;
  if (i < n) p[i] = 0.0f;
}

__global__ __launch_bounds__(256)
void pool_acc_kernel(const float* __restrict__ xn, const int* __restrict__ pid,
                     float* __restrict__ sums, float* __restrict__ counts) {
  const int bs = blockIdx.x;
  const int b  = bs / S_;
  const int p  = pid[bs];
  const int t  = threadIdx.x;
  float* dst = sums + (size_t)(b * P_ + p) * D_;
  const float* src = xn + (size_t)bs * D_;
  atomicAdd(&dst[t],       src[t]);
  atomicAdd(&dst[t + 256], src[t + 256]);
  if (t == 0) atomicAdd(&counts[b * P_ + p], 1.0f);
}

__global__ __launch_bounds__(256)
void pool_mean_kernel(const float* __restrict__ sums, const float* __restrict__ counts,
                      __bf16* __restrict__ means_bf) {
  const int bp = blockIdx.x;
  const int t  = threadIdx.x;
  float inv = 1.0f / fmaxf(counts[bp], 1.0f);
  means_bf[(size_t)bp * D_ + t]       = (__bf16)(sums[(size_t)bp * D_ + t]       * inv);
  means_bf[(size_t)bp * D_ + t + 256] = (__bf16)(sums[(size_t)bp * D_ + t + 256] * inv);
}

extern "C" void kernel_launch(void* const* d_in, const int* in_sizes, int n_in,
                              void* d_out, int out_size, void* d_ws, size_t ws_size,
                              hipStream_t stream) {
  const int*   tok   = (const int*)d_in[0];
  const int*   pid   = (const int*)d_in[1];
  const float* emb   = (const float*)d_in[2];
  const float* pos   = (const float*)d_in[3];
  const float* Wqkv  = (const float*)d_in[4];
  const float* bqkv  = (const float*)d_in[5];
  const float* Wo    = (const float*)d_in[6];
  const float* bo    = (const float*)d_in[7];
  const float* ln1g  = (const float*)d_in[8];
  const float* ln1b  = (const float*)d_in[9];
  const float* ln2g  = (const float*)d_in[10];
  const float* ln2b  = (const float*)d_in[11];
  const float* W1    = (const float*)d_in[12];
  const float* b1    = (const float*)d_in[13];
  const float* W2    = (const float*)d_in[14];
  const float* b2    = (const float*)d_in[15];
  const float* lnfg  = (const float*)d_in[16];
  const float* lnfb  = (const float*)d_in[17];
  const float* Wproj = (const float*)d_in[18];
  const float* bproj = (const float*)d_in[19];

  char* ws = (char*)d_ws;
  size_t off = 0;
  auto alloc = [&](size_t bytes) -> void* {
    void* p = ws + off;
    off += (bytes + 255) & ~(size_t)255;
    return p;
  };
  const int M = B_ * S_;
  float*  x       = (float*)alloc((size_t)M * D_ * 4);
  float*  y       = (float*)alloc((size_t)M * D_ * 4);
  __bf16* x_bf    = (__bf16*)alloc((size_t)M * D_ * 2);
  __bf16* qkv_bf  = (__bf16*)alloc((size_t)M * 3 * D_ * 2);
  __bf16* o_bf    = (__bf16*)alloc((size_t)M * D_ * 2);
  __bf16* h_bf    = (__bf16*)alloc((size_t)M * F_ * 2);
  float*  sums    = (float*)alloc((size_t)B_ * P_ * D_ * 4);
  float*  cnts    = (float*)alloc((size_t)B_ * P_ * 4);
  __bf16* mns_bf  = (__bf16*)alloc((size_t)B_ * P_ * D_ * 2);
  __bf16* wqkv_t  = (__bf16*)alloc((size_t)L_ * 3 * D_ * D_ * 2);
  __bf16* wo_t    = (__bf16*)alloc((size_t)L_ * D_ * D_ * 2);
  __bf16* w1_t    = (__bf16*)alloc((size_t)L_ * F_ * D_ * 2);
  __bf16* w2_t    = (__bf16*)alloc((size_t)L_ * D_ * F_ * 2);
  __bf16* wproj_t = (__bf16*)alloc((size_t)G_ * D_ * 2);
  float*  bqkv_s  = (float*)alloc((size_t)L_ * 3 * D_ * 4);

  // One-time weight transpose/convert to bf16 [N,K] (K-section of Wqkv scaled 1/8)
  for (int l = 0; l < L_; ++l) {
    wconv_kernel<<<dim3((3 * D_) / 32, D_ / 32), 256, 0, stream>>>(
        Wqkv + (size_t)l * D_ * 3 * D_, wqkv_t + (size_t)l * 3 * D_ * D_,
        D_, 3 * D_, D_, 2 * D_, 0.125f);
    wconv_kernel<<<dim3(D_ / 32, D_ / 32), 256, 0, stream>>>(
        Wo + (size_t)l * D_ * D_, wo_t + (size_t)l * D_ * D_, D_, D_, 0, 0, 1.0f);
    wconv_kernel<<<dim3(F_ / 32, D_ / 32), 256, 0, stream>>>(
        W1 + (size_t)l * D_ * F_, w1_t + (size_t)l * F_ * D_, D_, F_, 0, 0, 1.0f);
    wconv_kernel<<<dim3(D_ / 32, F_ / 32), 256, 0, stream>>>(
        W2 + (size_t)l * F_ * D_, w2_t + (size_t)l * D_ * F_, F_, D_, 0, 0, 1.0f);
  }
  wconv_kernel<<<dim3(G_ / 32, D_ / 32), 256, 0, stream>>>(
      Wproj, wproj_t, D_, G_, 0, 0, 1.0f);
  bscale_kernel<<<(L_ * 3 * D_ + 255) / 256, 256, 0, stream>>>(bqkv, bqkv_s, L_ * 3 * D_);

  embed_kernel<<<M, 256, 0, stream>>>(tok, emb, pos, x, x_bf);

  for (int l = 0; l < L_; ++l) {
    gemm_kernel<<<dim3((3 * D_) / 256, M / 64), 256, 0, stream>>>(
        x_bf, wqkv_t + (size_t)l * 3 * D_ * D_, bqkv_s + (size_t)l * 3 * D_, nullptr,
        nullptr, qkv_bf, M, 3 * D_, D_, 0);
    attn_kernel<<<dim3(S_ / 128, H_, B_), 256, 0, stream>>>(qkv_bf, o_bf);
    gemm_kernel<<<dim3(D_ / 256, M / 64), 256, 0, stream>>>(
        o_bf, wo_t + (size_t)l * D_ * D_, bo + (size_t)l * D_, x,
        y, nullptr, M, D_, D_, 0);
    ln_kernel<<<M, 256, 0, stream>>>(y, ln1g + (size_t)l * D_, ln1b + (size_t)l * D_, x, x_bf);
    gemm_kernel<<<dim3(F_ / 256, M / 64), 256, 0, stream>>>(
        x_bf, w1_t + (size_t)l * F_ * D_, b1 + (size_t)l * F_, nullptr,
        nullptr, h_bf, M, F_, D_, 1);
    gemm_kernel<<<dim3(D_ / 256, M / 64), 256, 0, stream>>>(
        h_bf, w2_t + (size_t)l * D_ * F_, b2 + (size_t)l * D_, x,
        y, nullptr, M, D_, F_, 0);
    ln_kernel<<<M, 256, 0, stream>>>(y, ln2g + (size_t)l * D_, ln2b + (size_t)l * D_, x, x_bf);
  }

  ln_kernel<<<M, 256, 0, stream>>>(x, lnfg, lnfb, y, nullptr);

  const int nsum = B_ * P_ * D_;
  zero_kernel<<<(nsum + 255) / 256, 256, 0, stream>>>(sums, nsum);
  zero_kernel<<<(B_ * P_ + 255) / 256, 256, 0, stream>>>(cnts, B_ * P_);
  pool_acc_kernel<<<M, 256, 0, stream>>>(y, pid, sums, cnts);
  pool_mean_kernel<<<B_ * P_, 256, 0, stream>>>(sums, cnts, mns_bf);

  gemm_kernel<<<dim3(G_ / 256, (B_ * P_) / 64), 256, 0, stream>>>(
      mns_bf, wproj_t, bproj, nullptr, (float*)d_out, nullptr, B_ * P_, G_, D_, 0);
}
